// LEGOTransformer_45749991637217
// MI455X (gfx1250) — compile-verified, matrix operand order checked by reading, not run on hardware
//
#include <hip/hip_runtime.h>
#include <math.h>

typedef __attribute__((ext_vector_type(16))) __bf16 v16bf;
typedef __attribute__((ext_vector_type(8)))  __bf16 v8bf;
typedef __attribute__((ext_vector_type(4)))  __bf16 v4bf;
typedef __attribute__((ext_vector_type(2)))  __bf16 v2bf;
typedef __attribute__((ext_vector_type(8)))  float  v8f;

#define T_TOK   4096
#define DIM     512
#define FFN_DIM 2048
#define VOCAB   32000

__device__ __forceinline__ float gelu_f(float x) {
    float x3 = x * x * x;
    return 0.5f * x * (1.0f + tanhf(0.7978845608028654f * (x + 0.044715f * x3)));
}

__device__ __forceinline__ void atomicMaxF(float* addr, float v) {
    if (v >= 0.0f) atomicMax((int*)addr, __float_as_int(v));
    else           atomicMin((unsigned int*)addr, __float_as_uint(v));
}

__device__ __forceinline__ v16bf cat16(v8bf lo, v8bf hi) {
    return __builtin_shufflevector(lo, hi, 0, 1, 2, 3, 4, 5, 6, 7,
                                           8, 9, 10, 11, 12, 13, 14, 15);
}

// -------------------- embedding gather --------------------
__global__ __launch_bounds__(128) void gather_kernel(const int* __restrict__ x,
                                                     const float* __restrict__ emb,
                                                     float* __restrict__ h) {
    int t = blockIdx.x;
    const float4* src = (const float4*)(emb + (size_t)x[t] * DIM);
    float4* dst = (float4*)(h + (size_t)t * DIM);
    dst[threadIdx.x] = src[threadIdx.x];
}

// -------------------- layernorm: one wave per token --------------------
__global__ __launch_bounds__(256) void ln_kernel(const float* __restrict__ x,
                                                 const float* __restrict__ s,
                                                 const float* __restrict__ b,
                                                 float* __restrict__ y) {
    int wave = threadIdx.x >> 5, lane = threadIdx.x & 31;
    int t = blockIdx.x * 8 + wave;
    const float* xr = x + (size_t)t * DIM;
    float sum = 0.0f, sum2 = 0.0f;
    #pragma unroll
    for (int i = lane; i < DIM; i += 32) { float v = xr[i]; sum += v; sum2 += v * v; }
    #pragma unroll
    for (int off = 16; off; off >>= 1) {
        sum  += __shfl_xor(sum,  off, 32);
        sum2 += __shfl_xor(sum2, off, 32);
    }
    float m   = sum  * (1.0f / DIM);
    float var = sum2 * (1.0f / DIM) - m * m;
    float inv = rsqrtf(var + 1e-5f);
    float* yr = y + (size_t)t * DIM;
    #pragma unroll
    for (int i = lane; i < DIM; i += 32) yr[i] = (xr[i] - m) * inv * s[i] + b[i];
}

// ===========================================================================
// Shared-tile geometry for WMMA GEMMs:
//   block tile 32(M) x 256(N), K-step 32, 256 threads = 8 waves.
//   wave (wr = w>>2, wc = w&3) owns rows wr*16..+16, cols wc*64..+64
//   (four 16x16 wmma tiles, one shared A fragment).
// LDS layouts:
//   As[row][40 bf16]      : K contiguous per row (stride 80B, 16B-aligned chunks)
//   Bs2[n][20 v2bf pairs] : K-pairs contiguous per column (stride 80B)
// Fragment loads are 2x ds_load_b128 (A) and 2x ds_load_b128 per B frag.
// K-loop is software-pipelined: next tile's global loads issue while the
// current tile's WMMAs drain, hiding HBM/L2 latency behind the matrix pipe.
// ===========================================================================

__device__ __forceinline__ void frag_A(const __bf16 (*As)[40], int row, int half,
                                       v16bf& af) {
    v8bf lo = *(const v8bf*)&As[row][half * 8];
    v8bf hi = *(const v8bf*)&As[row][half * 8 + 16];
    af = cat16(lo, hi);
}
__device__ __forceinline__ void frag_B(const v2bf (*Bs2)[20], int n, int half,
                                       v16bf& bfv) {
    v8bf lo = *(const v8bf*)&Bs2[n][half * 4];
    v8bf hi = *(const v8bf*)&Bs2[n][8 + half * 4];
    bfv = cat16(lo, hi);
}

// -------------------- WMMA GEMM: C = epi(A[MxK] @ B[KxN] (+Cin)) ----------
template<int EPI>  // 0 = none, 1 = add residual, 2 = gelu
__global__ __launch_bounds__(256) void gemm_kernel(const float* __restrict__ A,
                                                   const float* __restrict__ B,
                                                   const float* __restrict__ Cin,
                                                   float* __restrict__ Cout,
                                                   int N, int K) {
    __shared__ __attribute__((aligned(16))) __bf16 As[32][40];
    __shared__ __attribute__((aligned(16))) v2bf   Bs2[256][20];
    const int tid  = threadIdx.x;
    const int wave = tid >> 5, lane = tid & 31;
    const int wr = wave >> 2, wc = wave & 3;
    const int half = lane >> 4, l16 = lane & 31 & 15;
    const int m0 = blockIdx.y * 32;
    const int n0 = blockIdx.x * 256;

    const int arow = tid >> 3,        acol = (tid & 7) * 4;   // A stage: float4 each
    const int bn2  = (tid & 127) * 2;                         // B stage: 2 cols
    const int bpb  = (tid >> 7) * 8;                          // 8 K-pairs

    const float* Aptr = A + (size_t)(m0 + arow) * K + acol;
    const float* Bptr = B + (size_t)(2 * bpb) * N + n0 + bn2;

    v8f acc[4] = {};

    // prologue: prefetch tile kk = 0
    float4 av = *(const float4*)Aptr;
    float2 be[8], bo[8];
    #pragma unroll
    for (int i = 0; i < 8; ++i) {
        const float* r0 = Bptr + (size_t)(2 * i) * N;
        be[i] = *(const float2*)r0;
        bo[i] = *(const float2*)(r0 + N);
    }

    for (int kk = 0; kk < K; kk += 32) {
        // drain staged registers into LDS
        v4bf ap = {(__bf16)av.x, (__bf16)av.y, (__bf16)av.z, (__bf16)av.w};
        *(v4bf*)&As[arow][acol] = ap;
        #pragma unroll
        for (int i = 0; i < 8; ++i) {
            v2bf p0 = {(__bf16)be[i].x, (__bf16)bo[i].x};
            v2bf p1 = {(__bf16)be[i].y, (__bf16)bo[i].y};
            Bs2[bn2][bpb + i]     = p0;
            Bs2[bn2 + 1][bpb + i] = p1;
        }
        __syncthreads();
        // prefetch tile kk+32 (latency hidden behind WMMAs below)
        if (kk + 32 < K) {
            av = *(const float4*)(Aptr + kk + 32);
            #pragma unroll
            for (int i = 0; i < 8; ++i) {
                const float* r0 = Bptr + (size_t)(kk + 32 + 2 * i) * N;
                be[i] = *(const float2*)r0;
                bo[i] = *(const float2*)(r0 + N);
            }
        }
        // compute
        v16bf af;
        frag_A(As, wr * 16 + l16, half, af);
        #pragma unroll
        for (int t_ = 0; t_ < 4; ++t_) {
            v16bf bfv;
            frag_B(Bs2, wc * 64 + t_ * 16 + l16, half, bfv);
            acc[t_] = __builtin_amdgcn_wmma_f32_16x16x32_bf16(
                false, af, false, bfv, (short)0, acc[t_], false, false);
        }
        __syncthreads();
    }

    #pragma unroll
    for (int t_ = 0; t_ < 4; ++t_) {
        const int col = n0 + wc * 64 + t_ * 16 + l16;
        #pragma unroll
        for (int r = 0; r < 8; ++r) {
            int row = m0 + wr * 16 + half * 8 + r;
            size_t i0 = (size_t)row * N + col;
            float v0 = acc[t_][r];
            if (EPI == 1) v0 += Cin[i0];
            if (EPI == 2) v0 = gelu_f(v0);
            Cout[i0] = v0;
        }
    }
}

// -------------------- head passes: logits = hb @ head^T --------------------
// PASS 0: per-row max -> rowmax (rows with exitblk < 0)
// PASS 1: per-row sum exp(l - max) -> sumexp
// PASS 2: store logits rows where exitblk[row] == bidx
template<int PASS>
__global__ __launch_bounds__(256) void head_pass_kernel(const float* __restrict__ A,
                                                        const float* __restrict__ Bh,
                                                        float* __restrict__ rowmax,
                                                        float* __restrict__ sumexp,
                                                        const int* __restrict__ exitblk,
                                                        float* __restrict__ out,
                                                        int bidx) {
    __shared__ __attribute__((aligned(16))) __bf16 As[32][40];
    __shared__ __attribute__((aligned(16))) v2bf   Bs2[256][20];
    __shared__ int sflag;
    const int tid  = threadIdx.x;
    const int wave = tid >> 5, lane = tid & 31;
    const int wr = wave >> 2, wc = wave & 3;
    const int half = lane >> 4, l16 = lane & 15;
    const int m0 = blockIdx.y * 32;
    const int n0 = blockIdx.x * 256;

    if (tid == 0) sflag = 0;
    __syncthreads();
    int probe = exitblk[m0 + (tid & 31)];
    bool want = (PASS == 2) ? (probe == bidx) : (probe < 0);
    if (want) sflag = 1;
    __syncthreads();
    if (!sflag) return;

    const int arow = tid >> 3, acol = (tid & 7) * 4;
    // head is [VOCAB][DIM]: K contiguous per row -> coalesced float4 per
    // 8-thread group walking one row (c = idx>>3, f4 = idx&7)
    const int hc = tid >> 3, hf4 = tid & 7;   // thread's column / float4 slot

    const float* Aptr = A + (size_t)(m0 + arow) * DIM + acol;
    const float* Hrow = Bh + (size_t)(n0 + hc) * DIM + hf4 * 4;

    v8f acc[4] = {};

    // prologue: prefetch tile kk = 0 (each thread: 8 columns spaced by 32)
    float4 av = *(const float4*)Aptr;
    float4 bv[8];
    #pragma unroll
    for (int i = 0; i < 8; ++i)
        bv[i] = *(const float4*)(Hrow + (size_t)(32 * i) * DIM);

    for (int kk = 0; kk < DIM; kk += 32) {
        v4bf ap = {(__bf16)av.x, (__bf16)av.y, (__bf16)av.z, (__bf16)av.w};
        *(v4bf*)&As[arow][acol] = ap;
        #pragma unroll
        for (int i = 0; i < 8; ++i) {
            v4bf bp = {(__bf16)bv[i].x, (__bf16)bv[i].y,
                       (__bf16)bv[i].z, (__bf16)bv[i].w};
            *(v4bf*)&Bs2[hc + 32 * i][2 * hf4] = bp;
        }
        __syncthreads();
        if (kk + 32 < DIM) {
            av = *(const float4*)(Aptr + kk + 32);
            #pragma unroll
            for (int i = 0; i < 8; ++i)
                bv[i] = *(const float4*)(Hrow + (size_t)(32 * i) * DIM + kk + 32);
        }
        v16bf af;
        frag_A(As, wr * 16 + l16, half, af);
        #pragma unroll
        for (int t_ = 0; t_ < 4; ++t_) {
            v16bf bfv;
            frag_B(Bs2, wc * 64 + t_ * 16 + l16, half, bfv);
            acc[t_] = __builtin_amdgcn_wmma_f32_16x16x32_bf16(
                false, af, false, bfv, (short)0, acc[t_], false, false);
        }
        __syncthreads();
    }

    #pragma unroll
    for (int r = 0; r < 8; ++r) {
        int row = m0 + wr * 16 + half * 8 + r;
        if (PASS == 0) {
            float v = fmaxf(fmaxf(acc[0][r], acc[1][r]), fmaxf(acc[2][r], acc[3][r]));
            #pragma unroll
            for (int off = 8; off; off >>= 1) v = fmaxf(v, __shfl_xor(v, off, 16));
            if (l16 == 0) atomicMaxF(&rowmax[row], v);
        } else if (PASS == 1) {
            float mx = rowmax[row];
            float e = __expf(acc[0][r] - mx) + __expf(acc[1][r] - mx)
                    + __expf(acc[2][r] - mx) + __expf(acc[3][r] - mx);
            #pragma unroll
            for (int off = 8; off; off >>= 1) e += __shfl_xor(e, off, 16);
            if (l16 == 0) atomicAdd(&sumexp[row], e);
        } else {
            if (exitblk[row] == bidx) {
                size_t i0 = (size_t)row * VOCAB + n0 + wc * 64 + l16;
                out[i0]      = acc[0][r];
                out[i0 + 16] = acc[1][r];
                out[i0 + 32] = acc[2][r];
                out[i0 + 48] = acc[3][r];
            }
        }
    }
}

// -------------------- small helpers --------------------
__global__ void init_exit_kernel(int* eb) {
    eb[blockIdx.x * blockDim.x + threadIdx.x] = -1;
}
__global__ void init_stats_kernel(float* mx, float* se) {
    int t = blockIdx.x * blockDim.x + threadIdx.x;
    mx[t] = __uint_as_float(0xFF800000u);  // -inf
    se[t] = 0.0f;
}
__global__ void decide_kernel(const float* __restrict__ sumexp, int* __restrict__ eb,
                              float thr, int bidx, int is_last) {
    int t = blockIdx.x * blockDim.x + threadIdx.x;
    if (eb[t] < 0) {
        float conf = 1.0f / sumexp[t];
        if (conf > thr || is_last) eb[t] = bidx;
    }
}

// -------------------- host orchestration --------------------
extern "C" void kernel_launch(void* const* d_in, const int* in_sizes, int n_in,
                              void* d_out, int out_size, void* d_ws, size_t ws_size,
                              hipStream_t stream) {
    const int*   x    = (const int*)  d_in[0];
    const float* emb  = (const float*)d_in[1];
    const float* head = (const float*)d_in[2];
    const float* Wv   = (const float*)d_in[3];
    const float* Wo   = (const float*)d_in[4];
    const float* W1   = (const float*)d_in[5];
    const float* W2   = (const float*)d_in[6];
    const float* ln1s = (const float*)d_in[7];
    const float* ln1b = (const float*)d_in[8];
    const float* ln2s = (const float*)d_in[9];
    const float* ln2b = (const float*)d_in[10];
    float* out = (float*)d_out;

    float* ws     = (float*)d_ws;
    float* h      = ws;                         // T x 512
    float* hn     = h   + (size_t)T_TOK * DIM;  // T x 512
    float* tmp    = hn  + (size_t)T_TOK * DIM;  // T x 512
    float* ffn    = tmp + (size_t)T_TOK * DIM;  // T x 2048
    float* rowmax = ffn + (size_t)T_TOK * FFN_DIM;
    float* sumexp = rowmax + T_TOK;
    int*   eblk   = (int*)(sumexp + T_TOK);

    const float THR[3] = {3.5e-5f, 4.0e-5f, 1.0f};

    gather_kernel<<<T_TOK, 128, 0, stream>>>(x, emb, h);
    init_exit_kernel<<<T_TOK / 256, 256, 0, stream>>>(eblk);

    dim3 gD(DIM / 256,     T_TOK / 32);   // N=512
    dim3 gF(FFN_DIM / 256, T_TOK / 32);   // N=2048
    dim3 gH(VOCAB / 256,   T_TOK / 32);   // N=32000

    int l = 0;
    for (int bi = 0; bi < 3; ++bi) {
        for (int j = 0; j < 2; ++j, ++l) {
            ln_kernel<<<T_TOK / 8, 256, 0, stream>>>(h, ln1s + l * DIM, ln1b + l * DIM, hn);
            gemm_kernel<0><<<gD, 256, 0, stream>>>(hn, Wv + (size_t)l * DIM * DIM,
                                                   nullptr, tmp, DIM, DIM);
            gemm_kernel<1><<<gD, 256, 0, stream>>>(tmp, Wo + (size_t)l * DIM * DIM,
                                                   h, h, DIM, DIM);
            ln_kernel<<<T_TOK / 8, 256, 0, stream>>>(h, ln2s + l * DIM, ln2b + l * DIM, hn);
            gemm_kernel<2><<<gF, 256, 0, stream>>>(hn, W1 + (size_t)l * DIM * FFN_DIM,
                                                   nullptr, ffn, FFN_DIM, DIM);
            gemm_kernel<1><<<gD, 256, 0, stream>>>(ffn, W2 + (size_t)l * FFN_DIM * DIM,
                                                   h, h, DIM, FFN_DIM);
        }
        init_stats_kernel<<<T_TOK / 256, 256, 0, stream>>>(rowmax, sumexp);
        head_pass_kernel<0><<<gH, 256, 0, stream>>>(h, head, rowmax, sumexp, eblk, out, bi);
        head_pass_kernel<1><<<gH, 256, 0, stream>>>(h, head, rowmax, sumexp, eblk, out, bi);
        decide_kernel<<<T_TOK / 256, 256, 0, stream>>>(sumexp, eblk, THR[bi], bi, bi == 2);
        head_pass_kernel<2><<<gH, 256, 0, stream>>>(h, head, rowmax, sumexp, eblk, out, bi);
    }
}
